// MaskRCNN_3564822856155
// MI455X (gfx1250) — compile-verified, matrix operand order checked by reading, not run on hardware
//
#include <hip/hip_runtime.h>
#include <stdint.h>

#define NBOX 4096
#define NCLS 5            // foreground classes (reference drops class 0)
#define SCORE_TH 0.05f
#define NMS_TH   0.3f
#define DET_LIM  100
#define IM_MAX   1023.0f

typedef __attribute__((ext_vector_type(16))) _Float16 v16h;
typedef __attribute__((ext_vector_type(8)))  float    v8f;

// ---------------------------------------------------------------------------
// Kernel 1: softmax over 6 classes + bbox delta decode (classes 1..5).
// Writes decoded boxes straight into d_out[(cc, i, 0..5)] (stride-7 layout)
// and per-class scores into ws_scores.
// ---------------------------------------------------------------------------
__global__ __launch_bounds__(256)
void decode_kernel(const float* __restrict__ props,    // (N,7): [0, lo3, hi3]
                   const float* __restrict__ deltas,   // (N,36)
                   const float* __restrict__ cls,      // (N,6)
                   float* __restrict__ out,            // (5,N,7)
                   float* __restrict__ ws_scores)      // (5,N)
{
    int i = blockIdx.x * blockDim.x + threadIdx.x;
    if (i >= NBOX) return;

    // softmax (max-subtracted, like jax.nn.softmax)
    float x[6];
    float m = -3.4e38f;
#pragma unroll
    for (int c = 0; c < 6; ++c) { x[c] = cls[i * 6 + c]; m = fmaxf(m, x[c]); }
    float sum = 0.f;
#pragma unroll
    for (int c = 0; c < 6; ++c) { x[c] = expf(x[c] - m); sum += x[c]; }

    float lo[3], sz[3], ctr[3];
#pragma unroll
    for (int k = 0; k < 3; ++k) {
        lo[k] = props[i * 7 + 1 + k];
        float hi = props[i * 7 + 4 + k];
        sz[k]  = hi - lo[k] + 1.0f;
        ctr[k] = lo[k] + 0.5f * sz[k];
    }

    for (int cc = 0; cc < NCLS; ++cc) {
        int c = cc + 1;
        float* o = out + ((size_t)cc * NBOX + i) * 7;
#pragma unroll
        for (int k = 0; k < 3; ++k) {
            float dc = deltas[i * 36 + c * 6 + k]     / 10.0f;   // BBOX_W ctr
            float ds = deltas[i * 36 + c * 6 + 3 + k] / 5.0f;    // BBOX_W size
            float pc = dc * sz[k] + ctr[k];
            float ps = expf(ds) * sz[k];
            float l = fminf(fmaxf(pc - 0.5f * ps,        0.f), IM_MAX);
            float h = fminf(fmaxf(pc + 0.5f * ps - 1.0f, 0.f), IM_MAX);
            o[k]     = l;
            o[3 + k] = h;
        }
        ws_scores[(size_t)cc * NBOX + i] = x[c] / sum;
    }
}

// ---------------------------------------------------------------------------
// Kernel 2: per-class bitonic sort of (score desc, idx asc) in LDS,
// then gather boxes into sorted order. One 1024-thread workgroup per class.
// Total order on (score,idx) => bit-identical to stable argsort(-score).
// ---------------------------------------------------------------------------
__global__ __launch_bounds__(1024)
void sort_kernel(const float* __restrict__ ws_scores,  // (5,N)
                 const float* __restrict__ out,        // (5,N,7) decoded boxes
                 float* __restrict__ g_sscore,         // (5,N) sorted scores
                 int*   __restrict__ g_sidx,           // (5,N) orig index
                 float* __restrict__ g_sboxes)         // (5,N,6) sorted boxes
{
    __shared__ float key[NBOX];
    __shared__ int   idx[NBOX];
    const int tid = threadIdx.x;
    const int cc  = blockIdx.x;

    for (int r = tid; r < NBOX; r += 1024) {
        key[r] = ws_scores[(size_t)cc * NBOX + r];
        idx[r] = r;
    }
    __syncthreads();

    for (int k = 2; k <= NBOX; k <<= 1) {
        for (int j = k >> 1; j > 0; j >>= 1) {
            for (int ii = tid; ii < NBOX; ii += 1024) {
                int p = ii ^ j;
                if (p > ii) {
                    bool asc = ((ii & k) == 0);
                    float ka = key[ii], kb = key[p];
                    int   ia = idx[ii], ib = idx[p];
                    // "a before b": larger score first, tie -> smaller index
                    bool aFirst = (ka > kb) || (ka == kb && ia < ib);
                    bool doSwap = asc ? !aFirst : aFirst;
                    if (doSwap) {
                        key[ii] = kb; key[p] = ka;
                        idx[ii] = ib; idx[p] = ia;
                    }
                }
            }
            __syncthreads();
        }
    }

    for (int r = tid; r < NBOX; r += 1024) {
        int oi = idx[r];
        g_sscore[(size_t)cc * NBOX + r] = key[r];
        g_sidx  [(size_t)cc * NBOX + r] = oi;
        const float* bp = out + ((size_t)cc * NBOX + oi) * 7;
        float* dp = g_sboxes + ((size_t)cc * NBOX + r) * 6;
#pragma unroll
        for (int q = 0; q < 6; ++q) dp[q] = bp[q];
    }
}

// ---------------------------------------------------------------------------
// Kernel 3: greedy 3-D NMS, one 1024-thread workgroup per class.
// The whole 96KB sorted box table is staged into LDS (320KB/WGP on CDNA5)
// using the gfx1250 async global->LDS DMA path (ASYNCcnt).
// Dynamic LDS: 4096*6 floats (boxes) + 4096 ints (keep) = 114688 B.
// ---------------------------------------------------------------------------
__global__ __launch_bounds__(1024)
void nms_kernel(const float* __restrict__ g_sscore,
                const int*   __restrict__ g_sidx,
                const float* __restrict__ g_sboxes,
                float* __restrict__ g_kept)            // (5,N) original order
{
    extern __shared__ __align__(16) unsigned char smem[];
    float* bx   = (float*)smem;                              // NBOX*6 floats
    int*   keep = (int*)(smem + (size_t)NBOX * 6 * 4);       // NBOX ints

    const int tid = threadIdx.x;
    const int cc  = blockIdx.x;

    // ---- CDNA5 async global->LDS staging of the sorted box table ----
    {
        const unsigned char* g = (const unsigned char*)(g_sboxes + (size_t)cc * NBOX * 6);
        unsigned lbase = (unsigned)(uintptr_t)(void*)bx;     // LDS aperture: low 32b = LDS offset
        const int bytes = NBOX * 6 * 4;                      // 98304 = 6144 x b128
        for (int ofs = tid * 16; ofs < bytes; ofs += 1024 * 16) {
            unsigned la = lbase + (unsigned)ofs;
            unsigned long long ga = (unsigned long long)(uintptr_t)(g + ofs);
            asm volatile("global_load_async_to_lds_b128 %0, %1, off"
                         :: "v"(la), "v"(ga) : "memory");
        }
    }
    for (int r = tid; r < NBOX; r += 1024)
        keep[r] = (g_sscore[(size_t)cc * NBOX + r] > SCORE_TH) ? 1 : 0;

    asm volatile("s_wait_asynccnt 0" ::: "memory");          // our wave's DMAs landed
    __syncthreads();                                         // everyone's landed

    // ---- sequential greedy suppression (LDS broadcast of box i) ----
    for (int i = 0; i < NBOX - 1; ++i) {
        if (keep[i]) {                                       // uniform branch
            float l0 = bx[i*6+0], l1 = bx[i*6+1], l2 = bx[i*6+2];
            float h0 = bx[i*6+3], h1 = bx[i*6+4], h2 = bx[i*6+5];
            float vi = (h0-l0+1.f)*(h1-l1+1.f)*(h2-l2+1.f);
            for (int j = i + 1 + tid; j < NBOX; j += 1024) {
                if (!keep[j]) continue;                      // benign race: only 0s written
                const float* b = bx + j * 6;
                float w0 = fminf(h0, b[3]) - fmaxf(l0, b[0]) + 1.f;
                float w1 = fminf(h1, b[4]) - fmaxf(l1, b[1]) + 1.f;
                float w2 = fminf(h2, b[5]) - fmaxf(l2, b[2]) + 1.f;
                float inter = fmaxf(w0, 0.f) * fmaxf(w1, 0.f) * fmaxf(w2, 0.f);
                float vj = (b[3]-b[0]+1.f)*(b[4]-b[1]+1.f)*(b[5]-b[2]+1.f);
                float iou = inter / (vi + vj - inter);
                if (iou > NMS_TH) keep[j] = 0;
            }
        }
        __syncthreads();
    }

    // ---- scatter kept scores back to original order ----
    for (int r = tid; r < NBOX; r += 1024) {
        int oi = g_sidx[(size_t)cc * NBOX + r];
        g_kept[(size_t)cc * NBOX + oi] =
            keep[r] ? g_sscore[(size_t)cc * NBOX + r] : 0.0f;
    }
}

// ---------------------------------------------------------------------------
// Kernel 4: exact 100th-largest via binary search on float bit patterns
// (all values >= 0 so uint order == float order). The predicate count is
// done on the matrix pipe: 512 indicators per wave packed into a 16x32 f16
// A-tile, multiplied by B = ones with V_WMMA_F32_16X16X32_F16 (row sums,
// each replicated across 16 columns -> divide grand total by 16). All
// quantities are small integers, so f16 0/1 inputs + f32 accumulation are
// exact and the float LDS-atomic reduction is order-independent.
// ---------------------------------------------------------------------------
__global__ __launch_bounds__(1024)
void topk_kernel(const float* __restrict__ kept,  // (5*N) = 20480 = 40 x 512
                 float* __restrict__ out)         // (5,N,7)
{
    __shared__ float s_sum;
    const int tid  = threadIdx.x;
    const int wave = tid >> 5;
    const int lane = tid & 31;
    const int TOT  = NCLS * NBOX;                 // 20480
    const int CHUNKS = TOT / 512;                 // 40

    __builtin_prefetch(kept + tid, 0, 0);         // gfx1250 global_prefetch_b8

    v16h ones;
#pragma unroll
    for (int e = 0; e < 16; ++e) ones[e] = (_Float16)1.0f;

    unsigned lo = 0u, hi = 0xFFFFFFFFu;
    for (int it = 0; it < 33 && lo < hi; ++it) {  // uniform across block
        unsigned mid = (unsigned)((((unsigned long long)lo + (unsigned long long)hi) + 1ull) >> 1);
        if (tid == 0) s_sum = 0.0f;
        __syncthreads();

        v8f acc = {};
        for (int chunk = wave; chunk < CHUNKS; chunk += 32) {   // uniform per wave
            const float* p = kept + chunk * 512 + lane * 16;
            v16h a;
#pragma unroll
            for (int e = 0; e < 16; ++e)
                a[e] = (__float_as_uint(p[e]) >= mid) ? (_Float16)1.0f : (_Float16)0.0f;
            // D = A(16x32 indicators) x B(ones) + C : per-row indicator sums
            acc = __builtin_amdgcn_wmma_f32_16x16x32_f16(
                      false, a, false, ones, (short)0, acc, false, false);
        }
        float part = 0.0f;
#pragma unroll
        for (int e = 0; e < 8; ++e) part += acc[e];
        atomicAdd(&s_sum, part);                  // exact: integer-valued f32 adds
        __syncthreads();
        int cnt = (int)(s_sum * 0.0625f + 0.5f);  // /16 (column replication)
        __syncthreads();
        if (cnt >= DET_LIM) lo = mid; else hi = mid - 1;
    }

    float th = __uint_as_float(lo);
    for (int j = tid; j < TOT; j += 1024) {
        float v = kept[j];
        out[(size_t)j * 7 + 6] = (v >= th) ? v : 0.0f;
    }
}

// ---------------------------------------------------------------------------
extern "C" void kernel_launch(void* const* d_in, const int* in_sizes, int n_in,
                              void* d_out, int out_size, void* d_ws, size_t ws_size,
                              hipStream_t stream)
{
    const float* props  = (const float*)d_in[0];   // bbox_proposals (4096,7)
    const float* deltas = (const float*)d_in[1];   // bbox_pred      (4096,36)
    const float* cls    = (const float*)d_in[2];   // cls_score      (4096,6)
    float* out = (float*)d_out;                    // (5,4096,7)

    char* ws = (char*)d_ws;                        // all offsets 16B aligned
    const size_t SZ = (size_t)NCLS * NBOX;         // 20480
    float* ws_scores = (float*)(ws);                              // 5*N f32
    float* ws_ss     = (float*)(ws + SZ * 4);                     // sorted scores
    int*   ws_sidx   = (int*)  (ws + SZ * 8);                     // sorted->orig idx
    float* ws_sboxes = (float*)(ws + SZ * 12);                    // 5*N*6 f32
    float* ws_kept   = (float*)(ws + SZ * 12 + SZ * 24);          // 5*N f32
    // total workspace: 20480 * 40 = 819200 bytes

    decode_kernel<<<NBOX / 256, 256, 0, stream>>>(props, deltas, cls, out, ws_scores);
    sort_kernel  <<<NCLS, 1024, 0, stream>>>(ws_scores, out, ws_ss, ws_sidx, ws_sboxes);
    nms_kernel   <<<NCLS, 1024, (size_t)NBOX * 6 * 4 + (size_t)NBOX * 4, stream>>>
                 (ws_ss, ws_sidx, ws_sboxes, ws_kept);
    topk_kernel  <<<1, 1024, 0, stream>>>(ws_kept, out);
}
